// LSTMAutoParams_5454608466006
// MI455X (gfx1250) — compile-verified, matrix operand order checked by reading, not run on hardware
//
#include <hip/hip_runtime.h>
#include <hip/hip_bf16.h>
#include <stdint.h>

// Problem constants
#define HID     1536
#define GATES   6144        // 4*HID
#define IN      32
#define OUT     10
#define SUB     2048        // T * INPUT_SIZE sequential sub-steps
#define KC_H    48          // HID / 32 (K-chunks per hidden GEMV)
#define NTILES  384         // GATES / 16
#define NWG     48
#define NTHREADS 256
#define NWAVES  8           // 256 / 32 (wave32)

typedef __attribute__((ext_vector_type(16))) __bf16 v16bf;
typedef __attribute__((ext_vector_type(8)))  float  v8f;

// K index inside a 32-wide chunk for the 16-bit A/B WMMA operand layout
// (CDNA5 ISA: 16-bit A-Matrix 16x32). Element e of the v16bf
// (VGPR j = e/2, half h = e%2):
//   lanes 0-15 (hi=0): VGPR0..3 -> K 0..7, VGPR4..7 -> K 16..23
//   lanes 16-31 (hi=1): +8
__device__ __forceinline__ int klocal(int hi, int e) {
  int j = e >> 1, h = e & 1;
  int k = (j < 4) ? (2 * j + h) : (2 * (j - 4) + h + 16);
  return k + (hi ? 8 : 0);
}

__device__ __forceinline__ unsigned short f2bf(float f) {
  unsigned u = __float_as_uint(f);
  unsigned r = (u + 0x7FFFu + ((u >> 16) & 1u)) >> 16;   // round-to-nearest-even
  return (unsigned short)r;
}

__device__ __forceinline__ float sigmoidf_(float x) {
  return 1.0f / (1.0f + __expf(-x));
}

// -------------------------------------------------------------------------
// Prologue 1: reset persistent state (sync counters poisoned to 0xAA by the
// harness; x state must start as inputs[0]).
__global__ void init_state(const float* __restrict__ inputs,
                           float* __restrict__ wsx,
                           unsigned* __restrict__ sync) {
  int t = threadIdx.x;
  if (t < IN) wsx[t] = inputs[t];
  if (t < 2)  sync[t] = 0u;
}

// Prologue 2: fp32 -> bf16 weight conversion, pre-swizzled into the WMMA
// A-operand layout so the persistent kernel's A load per (tile, kchunk, lane)
// is one contiguous 32-byte block:
//   dst[(((tile*KC + kc)*32 + lane)*16) + e] = bf16( W[row, k] )
//   row = tile*16 + (lane&15), k = kc*32 + klocal(lane>>4, e)
__global__ void swizzle_w(const float* __restrict__ src,
                          unsigned short* __restrict__ dst, int K) {
  int KC = K >> 5;
  size_t total = (size_t)GATES * (size_t)K;
  for (size_t idx = (size_t)blockIdx.x * blockDim.x + threadIdx.x; idx < total;
       idx += (size_t)gridDim.x * blockDim.x) {
    int e    = (int)(idx & 15);
    int lane = (int)((idx >> 4) & 31);
    size_t blk = idx >> 9;            // tile*KC + kc
    int kc   = (int)(blk % KC);
    int tile = (int)(blk / KC);
    int hi   = lane >> 4;
    int row  = tile * 16 + (lane & 15);
    int k    = (kc << 5) + klocal(hi, e);
    dst[idx] = f2bf(src[(size_t)row * K + k]);
  }
}

// Grid-wide arrive/generation barrier (agent scope, s_sleep backoff).
__device__ __forceinline__ void grid_sync(unsigned* sync, unsigned* mygen) {
  __syncthreads();
  if (threadIdx.x == 0) {
    __threadfence();
    unsigned g = *mygen;
    unsigned arrived = atomicAdd(&sync[0], 1u) + 1u;
    if (arrived == gridDim.x) {
      atomicExch(&sync[0], 0u);
      __threadfence();
      atomicAdd(&sync[1], 1u);
    } else {
      while (__hip_atomic_load(&sync[1], __ATOMIC_RELAXED,
                               __HIP_MEMORY_SCOPE_AGENT) <= g) {
        __builtin_amdgcn_s_sleep(2);
      }
    }
  }
  __syncthreads();
  *mygen += 1;
  __threadfence();
}

// -------------------------------------------------------------------------
// Persistent kernel. 48 WGs x 8 wave32 = 384 waves = one 16-row gate tile
// per wave. h/c state replicated per-WG in LDS; gates + x go through global.
// 3 grid syncs per sub-step.
__global__ __launch_bounds__(NTHREADS, 1)
void lstm_persistent(const float* __restrict__ inputs,
                     const unsigned short* __restrict__ wih0,
                     const unsigned short* __restrict__ whh0,
                     const unsigned short* __restrict__ wih1,
                     const unsigned short* __restrict__ whh1,
                     const float* __restrict__ bih0, const float* __restrict__ bhh0,
                     const float* __restrict__ bih1, const float* __restrict__ bhh1,
                     const float* __restrict__ wlin, const float* __restrict__ blin,
                     const float* __restrict__ lookup,
                     float* wsx, float* gates0, float* gates1,
                     unsigned* sync, float* out) {
  __shared__ float h0s[HID], c0s[HID], h1s[HID], c1s[HID];
  __shared__ __align__(32) unsigned short ph0[HID];   // bf16 B-panel of h0
  __shared__ __align__(32) unsigned short ph1[HID];   // bf16 B-panel of h1
  __shared__ __align__(32) unsigned short px[IN];     // bf16 B-panel of x
  __shared__ float red[NTHREADS];
  __shared__ float lg[OUT];

  const int tid  = threadIdx.x;
  const int wave = tid >> 5;
  const int lane = tid & 31;
  const int hi   = lane >> 4;
  const int tile = blockIdx.x * NWAVES + wave;   // 0..383
  unsigned mygen = 0;

  for (int j = tid; j < HID; j += NTHREADS) {
    h0s[j] = 0.f; c0s[j] = 0.f; h1s[j] = 0.f; c1s[j] = 0.f;
    ph0[j] = 0;   ph1[j] = 0;
  }
  if (tid < IN) px[tid] = 0;
  __syncthreads();

  for (int s = 0; s < SUB; ++s) {
    // ---- Phase 1: gates0 = W_ih0 @ x + W_hh0 @ h0 + b ----
    if (tid < IN) px[tid] = f2bf(wsx[klocal(tid >> 4, tid & 15)]);
    __syncthreads();
    {
      v8f acc = {};
      v16bf a = *(const v16bf*)(wih0 + (((size_t)tile * 32 + lane) << 4));
      v16bf b = *(const v16bf*)(&px[hi << 4]);
      acc = __builtin_amdgcn_wmma_f32_16x16x32_bf16(false, a, false, b,
                                                    (short)0, acc, false, false);
      for (int kc = 0; kc < KC_H; ++kc) {
        v16bf aw = *(const v16bf*)(whh0 + ((((size_t)tile * KC_H + kc) * 32 + lane) << 4));
        v16bf bw = *(const v16bf*)(&ph0[(kc << 5) + (hi << 4)]);
        acc = __builtin_amdgcn_wmma_f32_16x16x32_bf16(false, aw, false, bw,
                                                      (short)0, acc, false, false);
      }
      // All D columns equal; lane 0 holds rows tile*16+0..7, lane 16 rows +8..15
      if ((lane & 15) == 0) {
        int rb = tile * 16 + hi * 8;
#pragma unroll
        for (int r = 0; r < 8; ++r) {
          int row = rb + r;
          gates0[row] = acc[r] + bih0[row] + bhh0[row];
        }
      }
    }
    grid_sync(sync, &mygen);

    // ---- Phase 2: local h0/c0 update, rebuild ph0, gates1 GEMVs ----
    for (int j = tid; j < HID; j += NTHREADS) {
      float ig = sigmoidf_(gates0[j]);
      float fg = sigmoidf_(gates0[j + HID]);
      float gg = tanhf(gates0[j + 2 * HID]);
      float og = sigmoidf_(gates0[j + 3 * HID]);
      float c  = fg * c0s[j] + ig * gg;
      c0s[j] = c;
      h0s[j] = og * tanhf(c);
    }
    __syncthreads();
    for (int p = tid; p < HID; p += NTHREADS) {
      int kc = p >> 5, hh = (p >> 4) & 1, e = p & 15;
      ph0[p] = f2bf(h0s[(kc << 5) + klocal(hh, e)]);
    }
    __syncthreads();
    {
      v8f acc = {};
      for (int kc = 0; kc < KC_H; ++kc) {
        v16bf aw = *(const v16bf*)(wih1 + ((((size_t)tile * KC_H + kc) * 32 + lane) << 4));
        v16bf bw = *(const v16bf*)(&ph0[(kc << 5) + (hi << 4)]);
        acc = __builtin_amdgcn_wmma_f32_16x16x32_bf16(false, aw, false, bw,
                                                      (short)0, acc, false, false);
      }
      for (int kc = 0; kc < KC_H; ++kc) {
        v16bf aw = *(const v16bf*)(whh1 + ((((size_t)tile * KC_H + kc) * 32 + lane) << 4));
        v16bf bw = *(const v16bf*)(&ph1[(kc << 5) + (hi << 4)]);
        acc = __builtin_amdgcn_wmma_f32_16x16x32_bf16(false, aw, false, bw,
                                                      (short)0, acc, false, false);
      }
      if ((lane & 15) == 0) {
        int rb = tile * 16 + hi * 8;
#pragma unroll
        for (int r = 0; r < 8; ++r) {
          int row = rb + r;
          gates1[row] = acc[r] + bih1[row] + bhh1[row];
        }
      }
    }
    grid_sync(sync, &mygen);

    // ---- Phase 3: local h1/c1 update, rebuild ph1; WG0 does head + sample ----
    for (int j = tid; j < HID; j += NTHREADS) {
      float ig = sigmoidf_(gates1[j]);
      float fg = sigmoidf_(gates1[j + HID]);
      float gg = tanhf(gates1[j + 2 * HID]);
      float og = sigmoidf_(gates1[j + 3 * HID]);
      float c  = fg * c1s[j] + ig * gg;
      c1s[j] = c;
      h1s[j] = og * tanhf(c);
    }
    __syncthreads();
    for (int p = tid; p < HID; p += NTHREADS) {
      int kc = p >> 5, hh = (p >> 4) & 1, e = p & 15;
      ph1[p] = f2bf(h1s[(kc << 5) + klocal(hh, e)]);
    }
    __syncthreads();

    if (blockIdx.x == 0) {
      for (int o = 0; o < OUT; ++o) {
        float part = 0.f;
        for (int j = tid; j < HID; j += NTHREADS) part += wlin[o * HID + j] * h1s[j];
        red[tid] = part;
        __syncthreads();
        for (int off = NTHREADS >> 1; off > 0; off >>= 1) {
          if (tid < off) red[tid] += red[tid + off];
          __syncthreads();
        }
        if (tid == 0) lg[o] = red[0] + blin[o];
        __syncthreads();
      }
      if (tid == 0) {
        float m = lg[0];
        for (int o = 1; o < OUT; ++o) m = fmaxf(m, lg[o]);
        float pr[OUT], sum = 0.f;
        for (int o = 0; o < OUT; ++o) { pr[o] = __expf(lg[o] - m); sum += pr[o]; }
        float inv = 1.f / sum;
        for (int o = 0; o < OUT; ++o) { pr[o] *= inv; out[SUB + s * OUT + o] = pr[o]; }
        // deterministic counter-hash categorical (stand-in for threefry key 42)
        unsigned long long z = 0x2545F4914F6CDD1DULL * 42ULL + (unsigned long long)s;
        z += 0x9E3779B97F4A7C15ULL;
        z = (z ^ (z >> 30)) * 0xBF58476D1CE4E5B9ULL;
        z = (z ^ (z >> 27)) * 0x94D049BB133111EBULL;
        z ^= (z >> 31);
        float u = (float)(z >> 40) * (1.0f / 16777216.0f);
        int action = OUT - 1;
        float cum = 0.f;
        for (int o = 0; o < OUT; ++o) { cum += pr[o]; if (u < cum) { action = o; break; } }
        out[s] = (float)action;
        // feedback: new_x = x scaled at position j (scale=1 when j==0);
        // next step uses fresh inputs[i] when (s+1)%32 == 0
        int j = s & 31;
        if (s + 1 < SUB) {
          if (((s + 1) & 31) == 0) {
            const float* inp = inputs + (size_t)((s + 1) >> 5) * IN;
            for (int k = 0; k < IN; ++k) wsx[k] = inp[k];
          } else if (j != 0) {
            wsx[j] = wsx[j] * lookup[action];
          }
        }
      }
    }
    grid_sync(sync, &mygen);
  }
}

// -------------------------------------------------------------------------
extern "C" void kernel_launch(void* const* d_in, const int* in_sizes, int n_in,
                              void* d_out, int out_size, void* d_ws, size_t ws_size,
                              hipStream_t stream) {
  (void)in_sizes; (void)n_in; (void)out_size;
  const float* inputs = (const float*)d_in[0];
  const float* W_ih0  = (const float*)d_in[3];
  const float* W_hh0  = (const float*)d_in[4];
  const float* b_ih0  = (const float*)d_in[5];
  const float* b_hh0  = (const float*)d_in[6];
  const float* W_ih1  = (const float*)d_in[7];
  const float* W_hh1  = (const float*)d_in[8];
  const float* b_ih1  = (const float*)d_in[9];
  const float* b_hh1  = (const float*)d_in[10];
  const float* W_lin  = (const float*)d_in[11];
  const float* b_lin  = (const float*)d_in[12];
  const float* lookup = (const float*)d_in[13];

  char* ws = (char*)d_ws;
  size_t off = 0;
  auto alloc = [&](size_t bytes) -> void* {
    off = (off + 255) & ~(size_t)255;
    void* p = ws + off;
    off += bytes;
    return p;
  };
  unsigned* sync   = (unsigned*)alloc(2 * sizeof(unsigned));
  float* wsx       = (float*)alloc(IN * sizeof(float));
  float* gates0    = (float*)alloc(GATES * sizeof(float));
  float* gates1    = (float*)alloc(GATES * sizeof(float));
  unsigned short* wih0sw = (unsigned short*)alloc((size_t)GATES * IN  * 2);
  unsigned short* whh0sw = (unsigned short*)alloc((size_t)GATES * HID * 2);
  unsigned short* wih1sw = (unsigned short*)alloc((size_t)GATES * HID * 2);
  unsigned short* whh1sw = (unsigned short*)alloc((size_t)GATES * HID * 2);
  if (off > ws_size) return;  // workspace too small; nothing safe to do

  init_state<<<1, 64, 0, stream>>>(inputs, wsx, sync);
  swizzle_w<<<768,  256, 0, stream>>>(W_ih0, wih0sw, IN);
  swizzle_w<<<4096, 256, 0, stream>>>(W_hh0, whh0sw, HID);
  swizzle_w<<<4096, 256, 0, stream>>>(W_ih1, wih1sw, HID);
  swizzle_w<<<4096, 256, 0, stream>>>(W_hh1, whh1sw, HID);
  lstm_persistent<<<NWG, NTHREADS, 0, stream>>>(
      inputs, wih0sw, whh0sw, wih1sw, whh1sw,
      b_ih0, b_hh0, b_ih1, b_hh1, W_lin, b_lin, lookup,
      wsx, gates0, gates1, sync, (float*)d_out);
}
